// OT_loss_12017318494251
// MI455X (gfx1250) — compile-verified, matrix operand order checked by reading, not run on hardware
//
#include <hip/hip_runtime.h>
#include <math.h>

typedef __attribute__((ext_vector_type(16))) _Float16 v16h;
typedef __attribute__((ext_vector_type(8)))  float    v8f;

#define B_   8
#define N_   16
#define M_   4
#define C_   2
#define HW_  65536
#define GAMMA_ 0.5f
#define BETA_  1.0f
#define EPS_   1e-8f

// workspace layout (floats)
#define WS_INTER 0      // B*N*M = 512
#define WS_SSUM  512    // B*N   = 128
#define WS_GCNT  640    // B*M   = 32

// ---------------------------------------------------------------------------
// Kernel 1: per batch b, compute via WMMA (f16 in, f32 acc):
//   D[n][m]   = sum_h sample[b,n,1,h] * (gt[b,m,h]==1)     (m = 0..3)
//   D[n][4]   = sum_h sample[b,n,1,h]                      (ones column)
//   gcnt[m]   = #{h : gt[b,m,h]==1}
// 8 waves per block, each wave owns a K-slice of HW/8, K unrolled x2 with
// two independent accumulators (pipelined WMMAs, no D->C serial hazard per
// iteration); LDS tree-reduce across waves.
// A-fragment layout (16-bit A 16x32): lane = row (lane&15), K-half = lane>>4;
// halves [0..7] = K base0..base0+7, halves [8..15] = K base0+16..base0+23,
// base0 = (lane>>4)*8. B-fragment mirrors this with lane&15 = column.
// ---------------------------------------------------------------------------
__global__ __launch_bounds__(256) void ot_cost_wmma(const int* __restrict__ gt,
                                                    const float* __restrict__ sample,
                                                    float* __restrict__ ws) {
  const int b     = blockIdx.x;
  const int wave  = threadIdx.x >> 5;
  const int lane  = threadIdx.x & 31;
  const int ln    = lane & 15;     // A row index / B column index
  const int kh    = lane >> 4;     // K half
  const int base0 = kh * 8;

  __shared__ float s_part[8][16][5];
  __shared__ float s_gcnt[8][4];

  const float* arow = sample + ((size_t)(b * N_ + ln) * C_ + 1) * (size_t)HW_;
  const int*   grow = gt + (size_t)(b * M_ + (ln < M_ ? ln : 0)) * (size_t)HW_;
  const bool   is_gt_lane = (ln < M_);

  // Loop-invariant B fragment for non-gt lanes: ones column at ln==M_,
  // zeros for ln>M_. Hoisted so the loop has a single divergent branch.
  v16h bf_base = {};
  if (ln == M_) {
#pragma unroll
    for (int i = 0; i < 16; ++i) bf_base[i] = (_Float16)1.0f;
  }

  v8f acc0 = {};
  v8f acc1 = {};
  int cnt = 0;

  const int k_begin = wave * (HW_ / 8);
  const int k_end   = k_begin + (HW_ / 8);

  for (int k0 = k_begin; k0 < k_end; k0 += 64) {
    // ---- two A fragments (K chunk k0 and k0+32) ----
    const float* ap = arow + k0 + base0;
    const float4 a0 = *(const float4*)(ap + 0);
    const float4 a1 = *(const float4*)(ap + 4);
    const float4 a2 = *(const float4*)(ap + 16);
    const float4 a3 = *(const float4*)(ap + 20);
    const float4 a4 = *(const float4*)(ap + 32);
    const float4 a5 = *(const float4*)(ap + 36);
    const float4 a6 = *(const float4*)(ap + 48);
    const float4 a7 = *(const float4*)(ap + 52);
    __builtin_prefetch(ap + 64, 0, 3);

    float af[32] = {a0.x, a0.y, a0.z, a0.w, a1.x, a1.y, a1.z, a1.w,
                    a2.x, a2.y, a2.z, a2.w, a3.x, a3.y, a3.z, a3.w,
                    a4.x, a4.y, a4.z, a4.w, a5.x, a5.y, a5.z, a5.w,
                    a6.x, a6.y, a6.z, a6.w, a7.x, a7.y, a7.z, a7.w};
    v16h aA, aB;
#pragma unroll
    for (int i = 0; i < 16; ++i) { aA[i] = (_Float16)af[i]; aB[i] = (_Float16)af[16 + i]; }

    // ---- two B fragments: one-hot(gt==1) on gt lanes, const elsewhere ----
    v16h bf0 = bf_base;
    v16h bf1 = bf_base;
    if (is_gt_lane) {
      const int* gp = grow + k0 + base0;
      const int4 g0 = *(const int4*)(gp + 0);
      const int4 g1 = *(const int4*)(gp + 4);
      const int4 g2 = *(const int4*)(gp + 16);
      const int4 g3 = *(const int4*)(gp + 20);
      const int4 g4 = *(const int4*)(gp + 32);
      const int4 g5 = *(const int4*)(gp + 36);
      const int4 g6 = *(const int4*)(gp + 48);
      const int4 g7 = *(const int4*)(gp + 52);
      int gi[32] = {g0.x, g0.y, g0.z, g0.w, g1.x, g1.y, g1.z, g1.w,
                    g2.x, g2.y, g2.z, g2.w, g3.x, g3.y, g3.z, g3.w,
                    g4.x, g4.y, g4.z, g4.w, g5.x, g5.y, g5.z, g5.w,
                    g6.x, g6.y, g6.z, g6.w, g7.x, g7.y, g7.z, g7.w};
#pragma unroll
      for (int i = 0; i < 16; ++i) {
        const int h0 = (gi[i] == 1);
        const int h1 = (gi[16 + i] == 1);
        bf0[i] = (_Float16)(float)h0;
        bf1[i] = (_Float16)(float)h1;
        cnt += h0 + h1;
      }
    }

    // Two independent accumulators -> the WMMAs can pipeline back-to-back.
    acc0 = __builtin_amdgcn_wmma_f32_16x16x32_f16(
        false, aA, false, bf0, (short)0, acc0, false, false);
    acc1 = __builtin_amdgcn_wmma_f32_16x16x32_f16(
        false, aB, false, bf1, (short)0, acc1, false, false);
  }

  v8f acc;
#pragma unroll
  for (int v = 0; v < 8; ++v) acc[v] = acc0[v] + acc1[v];

  // ---- stash this wave's partial tile: D row = v + kh*8, col = ln ----
  if (ln <= M_) {
#pragma unroll
    for (int v = 0; v < 8; ++v) s_part[wave][v + kh * 8][ln] = acc[v];
  }
  // gt==1 count: combine the two K-halves of each row m
  const int cnt2 = cnt + __shfl_xor(cnt, 16, 32);
  if (lane < M_) s_gcnt[wave][lane] = (float)cnt2;

  __syncthreads();

  // ---- deterministic cross-wave reduction to workspace ----
  const int t = threadIdx.x;
  if (t < 80) {
    const int row = t / 5, c5 = t % 5;
    float s = 0.0f;
#pragma unroll
    for (int w = 0; w < 8; ++w) s += s_part[w][row][c5];
    if (c5 < M_) ws[WS_INTER + (b * N_ + row) * M_ + c5] = s;
    else         ws[WS_SSUM + b * N_ + row] = s;
  } else if (t < 84) {
    const int m = t - 80;
    float s = 0.0f;
#pragma unroll
    for (int w = 0; w < 8; ++w) s += s_gcnt[w][m];
    ws[WS_GCNT + b * M_ + m] = s;
  }
}

// ---------------------------------------------------------------------------
// Kernel 2: tiny finalize. Lanes 0..7 each handle one batch:
//   cost -> greedy capacity-constrained coupling -> seg loss + KL.
// ---------------------------------------------------------------------------
__global__ __launch_bounds__(32) void ot_finalize(const float* __restrict__ ws,
                                                  const float* __restrict__ prob,
                                                  const float* __restrict__ prob_gt,
                                                  float* __restrict__ out) {
  __shared__ float s_seg[B_], s_kl[B_];
  const int b = threadIdx.x;

  if (b < B_) {
    float cost[N_][M_], ssum[N_], gcnt[M_];
    for (int n = 0; n < N_; ++n) ssum[n] = ws[WS_SSUM + b * N_ + n];
    for (int m = 0; m < M_; ++m) gcnt[m] = ws[WS_GCNT + b * M_ + m];
    for (int n = 0; n < N_; ++n)
      for (int m = 0; m < M_; ++m) {
        const float inter = ws[WS_INTER + (b * N_ + n) * M_ + m];
        const float uni   = ssum[n] + gcnt[m] - inter;
        cost[n][m] = 1.0f - (inter + 1.0f) / (uni + 1.0f);
      }

    // order_gt = stable argsort of per-column min
    float key[M_];
    for (int m = 0; m < M_; ++m) {
      float mn = cost[0][m];
      for (int n = 1; n < N_; ++n) mn = fminf(mn, cost[n][m]);
      key[m] = mn;
    }
    int order[M_];
    for (int m = 0; m < M_; ++m) order[m] = m;
    for (int i = 1; i < M_; ++i) {
      const int oi = order[i]; const float ki = key[oi];
      int j = i - 1;
      while (j >= 0 && key[order[j]] > ki) { order[j + 1] = order[j]; --j; }
      order[j + 1] = oi;
    }

    // greedy coupling (matches the lax.scan step exactly)
    float rowsums[N_]; float P[N_][M_];
    for (int n = 0; n < N_; ++n) { rowsums[n] = 0.0f; for (int m = 0; m < M_; ++m) P[n][m] = 0.0f; }
    const float mass = 1.0f / (float)M_;

    for (int t = 0; t < M_; ++t) {
      const int i = order[t];
      int ordj[N_];
      for (int n = 0; n < N_; ++n) ordj[n] = n;
      for (int a2 = 1; a2 < N_; ++a2) {          // stable argsort of column i
        const int oa = ordj[a2]; const float ka = cost[oa][i];
        int j = a2 - 1;
        while (j >= 0 && cost[ordj[j]][i] > ka) { ordj[j + 1] = ordj[j]; --j; }
        ordj[j + 1] = oa;
      }
      float cum = 0.0f;
      for (int j = 0; j < N_; ++j) {             // caps / exclusive-cumsum / clip
        const int r = ordj[j];
        const float cap  = GAMMA_ - rowsums[r];
        const float excl = cum;
        cum += cap;
        float a = mass - excl;
        a = fminf(fmaxf(a, 0.0f), cap);
        P[r][i] = a;
        rowsums[r] += a;
      }
    }

    for (int n = 0; n < N_; ++n)
      for (int m = 0; m < M_; ++m)
        P[n][m] *= prob_gt[b * M_ + m] * (float)M_;

    float seg = 0.0f, kl = 0.0f;
    for (int n = 0; n < N_; ++n) {
      float target = 0.0f;
      for (int m = 0; m < M_; ++m) { seg += P[n][m] * cost[n][m]; target += P[n][m]; }
      if (target > 0.0f)
        kl += target * (logf(target) - logf(prob[b * N_ + n] + EPS_));
    }
    s_seg[b] = seg;
    s_kl[b]  = kl;
  }
  __syncthreads();

  if (threadIdx.x == 0) {
    float seg = 0.0f, kl = 0.0f;
    for (int bb = 0; bb < B_; ++bb) { seg += s_seg[bb]; kl += s_kl[bb]; }
    seg /= (float)B_;
    kl  /= (float)(B_ * N_);
    out[0] = seg + BETA_ * kl;   // loss
    out[1] = seg;                // seg_loss
    out[2] = kl;                 // kl
  }
}

extern "C" void kernel_launch(void* const* d_in, const int* in_sizes, int n_in,
                              void* d_out, int out_size, void* d_ws, size_t ws_size,
                              hipStream_t stream) {
  const int*   gt      = (const int*)d_in[0];    // (B,M,H,W) int32
  const float* sample  = (const float*)d_in[1];  // (B,N,C,H,W) f32
  const float* prob    = (const float*)d_in[2];  // (B,N) f32
  const float* prob_gt = (const float*)d_in[3];  // (B,M) f32
  // d_in[4] = sample_shape (unused)
  float* ws  = (float*)d_ws;
  float* out = (float*)d_out;

  ot_cost_wmma<<<B_, 256, 0, stream>>>(gt, sample, ws);
  ot_finalize<<<1, 32, 0, stream>>>(ws, prob, prob_gt, out);
}
